// QMaskBlock_12747462934761
// MI455X (gfx1250) — matrix operand order repaired and checked by reading, not compiled, not run on hardware
//
#include <hip/hip_runtime.h>
#include <hip/hip_bf16.h>
#include <cstdint>
#include <cstddef>

#define BB    8
#define NIMG  1024
#define LQ    100
#define NTOT  1124
#define CC    768
#define HH    12
#define DD    64
#define HIDD  3072

typedef __attribute__((ext_vector_type(16))) __bf16 v16bf;
typedef __attribute__((ext_vector_type(8)))  __bf16 v8bf;
typedef __attribute__((ext_vector_type(8)))  float  v8f;

__device__ __forceinline__ float bf2f(__bf16 h) {
  unsigned short u = __builtin_bit_cast(unsigned short, h);
  unsigned int x = ((unsigned int)u) << 16;
  return __builtin_bit_cast(float, x);
}
__device__ __forceinline__ __bf16 f2bf(float f) {
  unsigned int u = __builtin_bit_cast(unsigned int, f);
  u += 0x7fffu + ((u >> 16) & 1u);           // round-to-nearest-even
  return __builtin_bit_cast(__bf16, (unsigned short)(u >> 16));
}
// A-matrix 16x32 bf16 fragment: per ISA layout, lane half hi supplies
// K chunks [hi*8, hi*8+7] and [16+hi*8, 16+hi*8+7] (two 16B loads).
__device__ __forceinline__ v16bf load_afrag(const __bf16* rowp, int hi) {
  union { v16bf v; v8bf h[2]; } u;
  u.h[0] = *(const v8bf*)(rowp + hi * 8);
  u.h[1] = *(const v8bf*)(rowp + 16 + hi * 8);
  return u.v;
}
// B-matrix 32x16 bf16 fragment: lane half hi supplies K chunks
// [hi*8, hi*8+7] and [16+hi*8, 16+hi*8+7] (two 16B loads), matching A.
__device__ __forceinline__ v16bf load_bfrag(const __bf16* rowp, int hi) {
  union { v16bf v; v8bf h[2]; } u;
  u.h[0] = *(const v8bf*)(rowp + hi * 8);
  u.h[1] = *(const v8bf*)(rowp + 16 + hi * 8);
  return u.v;
}
__device__ __forceinline__ v8f wmma_bf16(v16bf a, v16bf b, v8f c) {
  return __builtin_amdgcn_wmma_f32_16x16x32_bf16(false, a, false, b,
                                                 (short)0, c, false, false);
}

// ---------------------------------------------------------------- utilities
__global__ void qmb_cast_bf16(const float* __restrict__ src,
                              __bf16* __restrict__ dst, int n) {
  int i = blockIdx.x * 256 + threadIdx.x;
  if (i < n) dst[i] = f2bf(src[i]);
}

__global__ void qmb_gather_q(const float* __restrict__ x, float* __restrict__ q) {
  int i = blockIdx.x * 256 + threadIdx.x;
  if (i >= LQ * BB * CC) return;
  int c = i % CC, r = i / CC;
  int b = r / LQ, n = r % LQ;
  q[i] = x[((size_t)(b * NTOT + NIMG + n)) * CC + c];
}
__global__ void qmb_scatter_q(const float* __restrict__ q, float* __restrict__ x) {
  int i = blockIdx.x * 256 + threadIdx.x;
  if (i >= LQ * BB * CC) return;
  int c = i % CC, r = i / CC;
  int b = r / LQ, n = r % LQ;
  x[((size_t)(b * NTOT + NIMG + n)) * CC + c] = q[i];
}

// ----------------------------------------------------------------- layernorm
__global__ __launch_bounds__(256) void qmb_ln(const float* __restrict__ x,
                                              const float* __restrict__ w,
                                              const float* __restrict__ b,
                                              __bf16* __restrict__ y) {
  __shared__ float s_sum[256], s_sq[256];
  const int row = blockIdx.x, t = threadIdx.x;
  const float* xr = x + (size_t)row * CC;
  float v0 = xr[t], v1 = xr[t + 256], v2 = xr[t + 512];
  s_sum[t] = v0 + v1 + v2;
  s_sq[t] = v0 * v0 + v1 * v1 + v2 * v2;
  __syncthreads();
  for (int o = 128; o > 0; o >>= 1) {
    if (t < o) { s_sum[t] += s_sum[t + o]; s_sq[t] += s_sq[t + o]; }
    __syncthreads();
  }
  float mean = s_sum[0] * (1.f / CC);
  float var = s_sq[0] * (1.f / CC) - mean * mean;
  float rstd = rsqrtf(var + 1e-5f);
  __bf16* yr = y + (size_t)row * CC;
  yr[t]       = f2bf((v0 - mean) * rstd * w[t] + b[t]);
  yr[t + 256] = f2bf((v1 - mean) * rstd * w[t + 256] + b[t + 256]);
  yr[t + 512] = f2bf((v2 - mean) * rstd * w[t + 512] + b[t + 512]);
}

// ------------------------------------------------------------- WMMA GEMM
// OUT[M,N] = A[M,K] @ W[N,K]^T (+bias) (+epilogue)
// One wave computes a 32x64 output tile: 2 A-frags x 4 B-frags -> 8 WMMAs per
// 32-deep k-step (8 v8f accumulators). Raises L1/L2 reuse to ~21 FLOP/B.
// mode 0: bf16 store; mode 1: f32 store with residual add; mode 2: GELU->bf16
__global__ __launch_bounds__(256) void qmb_gemm(const __bf16* __restrict__ A,
                                                const __bf16* __restrict__ W,
                                                const float* __restrict__ bias,
                                                const float* __restrict__ resid,
                                                void* __restrict__ out,
                                                int M, int N, int K, int mode) {
  const int lane = threadIdx.x & 31, wave = threadIdx.x >> 5;
  const int NT = N >> 6, MT = (M + 31) >> 5;
  int tile = blockIdx.x * 8 + wave;
  if (tile >= MT * NT) return;
  int mt = tile / NT, nt = tile % NT;
  int hi = lane >> 4, l = lane & 15;
  int am0 = mt * 32 + l;      if (am0 >= M) am0 = M - 1;
  int am1 = mt * 32 + 16 + l; if (am1 >= M) am1 = M - 1;
  const __bf16* ap0 = A + (size_t)am0 * K;
  const __bf16* ap1 = A + (size_t)am1 * K;
  const __bf16* bp0 = W + (size_t)(nt * 64 + l) * K;
  const __bf16* bp1 = W + (size_t)(nt * 64 + 16 + l) * K;
  const __bf16* bp2 = W + (size_t)(nt * 64 + 32 + l) * K;
  const __bf16* bp3 = W + (size_t)(nt * 64 + 48 + l) * K;
  v8f acc[2][4] = {};
  for (int kk = 0; kk < K; kk += 32) {
    __builtin_prefetch(ap0 + kk + 128, 0, 1);
    __builtin_prefetch(bp0 + kk + 128, 0, 1);
    v16bf a0 = load_afrag(ap0 + kk, hi);
    v16bf a1 = load_afrag(ap1 + kk, hi);
    v16bf b0 = load_bfrag(bp0 + kk, hi);
    v16bf b1 = load_bfrag(bp1 + kk, hi);
    v16bf b2 = load_bfrag(bp2 + kk, hi);
    v16bf b3 = load_bfrag(bp3 + kk, hi);
    acc[0][0] = wmma_bf16(a0, b0, acc[0][0]);
    acc[0][1] = wmma_bf16(a0, b1, acc[0][1]);
    acc[0][2] = wmma_bf16(a0, b2, acc[0][2]);
    acc[0][3] = wmma_bf16(a0, b3, acc[0][3]);
    acc[1][0] = wmma_bf16(a1, b0, acc[1][0]);
    acc[1][1] = wmma_bf16(a1, b1, acc[1][1]);
    acc[1][2] = wmma_bf16(a1, b2, acc[1][2]);
    acc[1][3] = wmma_bf16(a1, b3, acc[1][3]);
  }
#pragma unroll
  for (int j = 0; j < 4; ++j) {
    int n = nt * 64 + j * 16 + l;
    float bn = bias ? bias[n] : 0.f;
#pragma unroll
    for (int i = 0; i < 2; ++i) {
#pragma unroll
      for (int r = 0; r < 8; ++r) {
        int m = mt * 32 + i * 16 + r + hi * 8;
        if (m >= M) continue;
        float v = acc[i][j][r] + bn;
        size_t idx = (size_t)m * N + n;
        if (mode == 0) {
          ((__bf16*)out)[idx] = f2bf(v);
        } else if (mode == 1) {
          ((float*)out)[idx] = resid[idx] + v;
        } else {
          float g = 0.5f * v * (1.f + erff(v * 0.70710678118654752f));
          ((__bf16*)out)[idx] = f2bf(g);
        }
      }
    }
  }
}

// ------------------------------------------------------------- attention
// One block = (b, h, 16-query tile). qkv rows [Bx*Nx, 3*CC] bf16, o [Bx*Nx, CC].
__global__ __launch_bounds__(128) void qmb_attn(const __bf16* __restrict__ qkv,
                                                __bf16* __restrict__ o,
                                                int Nx, int Nk, int QT) {
  const int KT = (Nk + 15) >> 4;   // 16-key score chunks (<=64)
  const int KC = (Nk + 31) >> 5;   // 32-key P@V chunks
  __shared__ __align__(32) __bf16 Pb[16][1024];  // unnormalized probs
  __shared__ __align__(32) __bf16 vt[64][32];    // transposed V chunk (d,key)
  __shared__ float maxc[16][64], sumc[16][64];
  __shared__ float maxg[16], invs[16];

  int id = blockIdx.x;
  int qt = id % QT, hh = (id / QT) % HH, b = id / (QT * HH);
  const int lane = threadIdx.x & 31, wave = threadIdx.x >> 5;
  const int hi = lane >> 4, l = lane & 15;
  const int RS = 3 * CC;

  int qrow = qt * 16 + l; if (qrow >= Nx) qrow = Nx - 1;
  const __bf16* qptr = qkv + (size_t)(b * Nx + qrow) * RS + hh * DD;

  // ---- S = scale * Q K^T, chunkwise exp with per-chunk max/sum ----
  for (int kt = wave; kt < KT; kt += 4) {
    int key = kt * 16 + l; int keyc = key < Nx ? key : Nx - 1;
    const __bf16* kptr = qkv + (size_t)(b * Nx + keyc) * RS + CC + hh * DD;
    v8f acc = {};
#pragma unroll
    for (int kk = 0; kk < DD; kk += 32) {
      v16bf af = load_afrag(qptr + kk, hi);
      v16bf bf = load_bfrag(kptr + kk, hi);
      acc = wmma_bf16(af, bf, acc);
    }
    int col = kt * 16 + l;
    float s[8], mx[8], e[8], sm[8];
#pragma unroll
    for (int r = 0; r < 8; ++r) s[r] = (col < Nk) ? acc[r] * 0.125f : -1e30f;
#pragma unroll
    for (int r = 0; r < 8; ++r) {
      float m = s[r];
      for (int d = 1; d < 16; d <<= 1) {
        float t = __shfl_xor(m, d, 32);
        m = m > t ? m : t;
      }
      mx[r] = m;
      e[r] = __expf(s[r] - m);
      float su = e[r];
      for (int d = 1; d < 16; d <<= 1) su += __shfl_xor(su, d, 32);
      sm[r] = su;
      Pb[r + hi * 8][col] = f2bf(e[r]);
    }
    if (l == 0) {
#pragma unroll
      for (int r = 0; r < 8; ++r) {
        maxc[r + hi * 8][kt] = mx[r];
        sumc[r + hi * 8][kt] = sm[r];
      }
    }
  }
  __syncthreads();
  // ---- global row max / inverse sum ----
  if (threadIdx.x < 16) {
    int r = threadIdx.x;
    float mg = -3.4e38f;
    for (int c = 0; c < KT; ++c) mg = fmaxf(mg, maxc[r][c]);
    float sum = 0.f;
    for (int c = 0; c < KT; ++c) sum += sumc[r][c] * __expf(maxc[r][c] - mg);
    maxg[r] = mg;
    invs[r] = 1.f / sum;
  }
  __syncthreads();
  // ---- rescale P chunks to global max, zero the 32-pad tail ----
  {
    int total = 16 * KT * 16;
    for (int i = threadIdx.x; i < total; i += 128) {
      int r = i / (KT * 16), c = i % (KT * 16);
      float sc = __expf(maxc[r][c >> 4] - maxg[r]);
      Pb[r][c] = f2bf(bf2f(Pb[r][c]) * sc);
    }
    int padlo = KT * 16, padhi = KC * 32;
    if (padhi > padlo) {
      int span = padhi - padlo;
      for (int i = threadIdx.x; i < 16 * span; i += 128)
        Pb[i / span][padlo + i % span] = f2bf(0.f);
    }
  }
  __syncthreads();
  // ---- O = P V : stage V transposed through LDS, wave w owns d cols w*16.. ----
  v8f oacc = {};
  for (int kc = 0; kc < KC; ++kc) {
    {
      int t = threadIdx.x;
      int keyl = t >> 2, db = (t & 3) * 16;
      int key = kc * 32 + keyl; if (key >= Nx) key = Nx - 1;
      const __bf16* vp = qkv + (size_t)(b * Nx + key) * RS + 2 * CC + hh * DD + db;
#pragma unroll
      for (int j = 0; j < 16; ++j) vt[db + j][keyl] = vp[j];
    }
    __syncthreads();
    {
      v16bf af = load_afrag(&Pb[l][kc * 32], hi);
      v16bf bf = load_bfrag(&vt[wave * 16 + l][0], hi);
      oacc = wmma_bf16(af, bf, oacc);
    }
    __syncthreads();
  }
  // ---- epilogue: normalize and store bf16 ----
#pragma unroll
  for (int r = 0; r < 8; ++r) {
    int m = r + hi * 8;
    int row = qt * 16 + m;
    if (row < Nx) {
      float v = oacc[r] * invs[m];
      o[(size_t)(b * Nx + row) * CC + hh * DD + wave * 16 + l] = f2bf(v);
    }
  }
}

// ----------------------------------------------------------------- host side
static inline void launch_gemm(const __bf16* A, const __bf16* W, const float* bias,
                               const float* resid, void* out, int M, int N, int K,
                               int mode, hipStream_t s) {
  int tiles = ((M + 31) / 32) * (N / 64);
  int blocks = (tiles + 7) / 8;
  qmb_gemm<<<blocks, 256, 0, s>>>(A, W, bias, resid, out, M, N, K, mode);
}

static void run_block(float* xb, int M, int Bx, int Nx, int Nk,
                      const __bf16* wqkv, const __bf16* wp,
                      const __bf16* wf1, const __bf16* wf2,
                      const float* n1w, const float* n1b, const float* pb,
                      const float* n2w, const float* n2b,
                      const float* f1b, const float* f2b,
                      __bf16* lnbuf, __bf16* qkvbuf, __bf16* obuf, __bf16* h1buf,
                      float* f2out, hipStream_t s) {
  int QT = (Nx + 15) / 16;
  qmb_ln<<<M, 256, 0, s>>>(xb, n1w, n1b, lnbuf);
  launch_gemm(lnbuf, wqkv, nullptr, nullptr, qkvbuf, M, 3 * CC, CC, 0, s);
  qmb_attn<<<Bx * HH * QT, 128, 0, s>>>(qkvbuf, obuf, Nx, Nk, QT);
  launch_gemm(obuf, wp, pb, xb, xb, M, CC, CC, 1, s);
  qmb_ln<<<M, 256, 0, s>>>(xb, n2w, n2b, lnbuf);
  launch_gemm(lnbuf, wf1, f1b, nullptr, h1buf, M, HIDD, CC, 2, s);
  launch_gemm(h1buf, wf2, f2b, xb, f2out, M, CC, HIDD, 1, s);
}

extern "C" void kernel_launch(void* const* d_in, const int* in_sizes, int n_in,
                              void* d_out, int out_size, void* d_ws, size_t ws_size,
                              hipStream_t stream) {
  (void)in_sizes; (void)n_in; (void)out_size; (void)ws_size;
  const float* x = (const float*)d_in[0];
  const float* qsa_n1w = (const float*)d_in[1];
  const float* qsa_n1b = (const float*)d_in[2];
  const float* qsa_qkvw = (const float*)d_in[3];
  const float* qsa_pw = (const float*)d_in[4];
  const float* qsa_pb = (const float*)d_in[5];
  const float* qsa_n2w = (const float*)d_in[6];
  const float* qsa_n2b = (const float*)d_in[7];
  const float* qsa_f1w = (const float*)d_in[8];
  const float* qsa_f1b = (const float*)d_in[9];
  const float* qsa_f2w = (const float*)d_in[10];
  const float* qsa_f2b = (const float*)d_in[11];
  const float* blk_n1w = (const float*)d_in[12];
  const float* blk_n1b = (const float*)d_in[13];
  const float* blk_qkvw = (const float*)d_in[14];
  const float* blk_pw = (const float*)d_in[15];
  const float* blk_pb = (const float*)d_in[16];
  const float* blk_n2w = (const float*)d_in[17];
  const float* blk_n2b = (const float*)d_in[18];
  const float* blk_f1w = (const float*)d_in[19];
  const float* blk_f1b = (const float*)d_in[20];
  const float* blk_f2w = (const float*)d_in[21];
  const float* blk_f2b = (const float*)d_in[22];

  const int M2 = BB * NTOT;   // 8992
  const int M1 = BB * LQ;     // 800

  // workspace layout (~128 MB)
  char* ws = (char*)d_ws;
  size_t off = 0;
  auto alloc = [&](size_t bytes) {
    size_t o = off;
    off = (off + bytes + 255) & ~(size_t)255;
    return o;
  };
  float*  xbuf   = (float*)(ws + alloc((size_t)M2 * CC * 4));
  float*  qbuf   = (float*)(ws + alloc((size_t)M1 * CC * 4));
  __bf16* lnbuf  = (__bf16*)(ws + alloc((size_t)M2 * CC * 2));
  size_t qkv_off = alloc((size_t)M2 * 3 * CC * 2);
  __bf16* qkvbuf = (__bf16*)(ws + qkv_off);
  __bf16* obuf   = (__bf16*)(ws + alloc((size_t)M2 * CC * 2));
  __bf16* h1buf  = (__bf16*)(ws + qkv_off);  // aliases qkv+obuf (dead by MLP time)
  __bf16* wq_qkv = (__bf16*)(ws + alloc((size_t)3 * CC * CC * 2));
  __bf16* wq_p   = (__bf16*)(ws + alloc((size_t)CC * CC * 2));
  __bf16* wq_f1  = (__bf16*)(ws + alloc((size_t)HIDD * CC * 2));
  __bf16* wq_f2  = (__bf16*)(ws + alloc((size_t)CC * HIDD * 2));
  __bf16* wb_qkv = (__bf16*)(ws + alloc((size_t)3 * CC * CC * 2));
  __bf16* wb_p   = (__bf16*)(ws + alloc((size_t)CC * CC * 2));
  __bf16* wb_f1  = (__bf16*)(ws + alloc((size_t)HIDD * CC * 2));
  __bf16* wb_f2  = (__bf16*)(ws + alloc((size_t)CC * HIDD * 2));

  // working copy of x (do not mutate inputs)
  hipMemcpyAsync(xbuf, x, (size_t)M2 * CC * 4, hipMemcpyDeviceToDevice, stream);

  auto cast = [&](const float* s, __bf16* d, int n) {
    qmb_cast_bf16<<<(n + 255) / 256, 256, 0, stream>>>(s, d, n);
  };
  cast(qsa_qkvw, wq_qkv, 3 * CC * CC);
  cast(qsa_pw,   wq_p,   CC * CC);
  cast(qsa_f1w,  wq_f1,  HIDD * CC);
  cast(qsa_f2w,  wq_f2,  CC * HIDD);
  cast(blk_qkvw, wb_qkv, 3 * CC * CC);
  cast(blk_pw,   wb_p,   CC * CC);
  cast(blk_f1w,  wb_f1,  HIDD * CC);
  cast(blk_f2w,  wb_f2,  CC * HIDD);

  // block 1: self-attention over the 100 query tokens only
  qmb_gather_q<<<(M1 * CC + 255) / 256, 256, 0, stream>>>(xbuf, qbuf);
  run_block(qbuf, M1, BB, LQ, LQ,
            wq_qkv, wq_p, wq_f1, wq_f2,
            qsa_n1w, qsa_n1b, qsa_pb, qsa_n2w, qsa_n2b, qsa_f1b, qsa_f2b,
            lnbuf, qkvbuf, obuf, h1buf, qbuf, stream);
  qmb_scatter_q<<<(M1 * CC + 255) / 256, 256, 0, stream>>>(qbuf, xbuf);

  // block 2: full sequence, last LEN_Q keys masked out of attention
  run_block(xbuf, M2, BB, NTOT, NIMG,
            wb_qkv, wb_p, wb_f1, wb_f2,
            blk_n1w, blk_n1b, blk_pb, blk_n2w, blk_n2b, blk_f1b, blk_f2b,
            lnbuf, qkvbuf, obuf, h1buf, (float*)d_out, stream);
}
